// Spectrogram_56410100465819
// MI455X (gfx1250) — matrix operand order repaired and checked by reading, not compiled, not run on hardware
//
#include <hip/hip_runtime.h>
#include <hip/hip_bf16.h>

// ---------------------------------------------------------------------------
// Spectrogram = two GEMMs on WMMA:
//   S = Wsin[1025x2048] * X[2048x20672],  C = Wcos * X,  out = S*S + C*C
// where X[k,n] = x_reflectpad[n*512 + k]  (implicit im2col: contiguous in k)
// ---------------------------------------------------------------------------

#define T_SAMPLES 10584000           // 44100*240
#define N_FFT     2048
#define PAD       1024
#define T_PAD     (T_SAMPLES + N_FFT) // 10,586,048
#define HOP       512
#define KDIM      2048
#define MBINS     1025
#define MPAD      1040                // 65 tiles of 16
#define NFRAMES   20672               // 1292 tiles of 16
#define MTILES    (MPAD / 16)         // 65
#define NPAIRS    (NFRAMES / 32)      // 646 (each wave does 2 N-tiles)

typedef __attribute__((ext_vector_type(16))) _Float16 v16h;
typedef __attribute__((ext_vector_type(8)))  float    v8f;
typedef __attribute__((ext_vector_type(4)))  unsigned int u32x4;

// ---------------- preprocessing ----------------

// reflect-pad (mode='reflect', excludes edge) + convert to f16
__global__ __launch_bounds__(256) void pad_convert_kernel(
    const float* __restrict__ x, _Float16* __restrict__ xh)
{
    int i = blockIdx.x * blockDim.x + threadIdx.x;
    if (i >= T_PAD) return;
    int j = i - PAD;
    if (j < 0) j = -j;
    if (j >= T_SAMPLES) j = 2 * T_SAMPLES - 2 - j;
    xh[i] = (_Float16)x[j];
}

// convert both weight matrices to f16, zero-pad rows 1025..1039
__global__ __launch_bounds__(256) void wconv_kernel(
    const float* __restrict__ wsin, const float* __restrict__ wcos,
    _Float16* __restrict__ whs, _Float16* __restrict__ whc)
{
    int idx = blockIdx.x * blockDim.x + threadIdx.x;
    if (idx >= MPAD * KDIM) return;
    int r = idx / KDIM;
    int c = idx - r * KDIM;
    float vs = 0.0f, vc = 0.0f;
    if (r < MBINS) {
        vs = wsin[r * KDIM + c];
        vc = wcos[r * KDIM + c];
    }
    whs[idx] = (_Float16)vs;
    whc[idx] = (_Float16)vc;
}

// ---------------- WMMA fragment loads ----------------
// A 16x32, contiguous k order chosen here: lane m=lane&15; hi=lane>>4 selects K half (+16);
// VGPR v holds K = 16*hi + 2v,2v+1 (same k order as B)
// => halves [k0+16*hi .. +15] of the weight row.
__device__ __forceinline__ v16h load_frag_a(const _Float16* __restrict__ row,
                                            int k0, int hi)
{
    const u32x4* p = (const u32x4*)(row + k0 + 16 * hi);
    union { v16h h; u32x4 u[2]; } r;
    r.u[0] = p[0];   // dwords d0..d0+3
    r.u[1] = p[1];   // dwords d0+4..d0+7
    return r.h;
}

// B 32x16, contiguous k order chosen here (as A): lane n=lane&15; hi selects K half (+16);
// VGPR v holds K = 16*hi + 2v,2v+1 => 16 consecutive halves of x_pad.
__device__ __forceinline__ v16h load_frag_b(const _Float16* __restrict__ base)
{
    const u32x4* p = (const u32x4*)base;
    union { v16h h; u32x4 u[2]; } r;
    r.u[0] = p[0];
    r.u[1] = p[1];
    return r.h;
}

// ---------------- main GEMM: one wave = 16 freq-bins x 32 frames ----------------
__global__ __launch_bounds__(256) void spect_wmma_kernel(
    const _Float16* __restrict__ whs, const _Float16* __restrict__ whc,
    const _Float16* __restrict__ xh, float* __restrict__ out)
{
    const int lane = threadIdx.x & 31;
    const int wave = blockIdx.x * (blockDim.x >> 5) + (threadIdx.x >> 5);
    if (wave >= MTILES * NPAIRS) return;   // wave-uniform exit: EXEC stays full

    const int mtile = wave / NPAIRS;
    const int npair = wave - mtile * NPAIRS;

    const int hi   = lane >> 4;
    const int mrow = mtile * 16 + (lane & 15);        // A-row this lane feeds
    const int n0   = npair * 32 + (lane & 15);        // B-col, first N-tile
    const int n1   = n0 + 16;                         // B-col, second N-tile

    const _Float16* __restrict__ arow_s = whs + (size_t)mrow * KDIM;
    const _Float16* __restrict__ arow_c = whc + (size_t)mrow * KDIM;
    const _Float16* __restrict__ b0p = xh + (size_t)n0 * HOP + 16 * hi;
    const _Float16* __restrict__ b1p = xh + (size_t)n1 * HOP + 16 * hi;

    v8f s0 = {}, c0 = {}, s1 = {}, c1 = {};

#pragma unroll 2
    for (int k0 = 0; k0 < KDIM; k0 += 32) {
        v16h as = load_frag_a(arow_s, k0, hi);
        v16h ac = load_frag_a(arow_c, k0, hi);
        v16h b0 = load_frag_b(b0p + k0);
        v16h b1 = load_frag_b(b1p + k0);
        s0 = __builtin_amdgcn_wmma_f32_16x16x32_f16(false, as, false, b0,
                                                    (short)0, s0, false, false);
        c0 = __builtin_amdgcn_wmma_f32_16x16x32_f16(false, ac, false, b0,
                                                    (short)0, c0, false, false);
        s1 = __builtin_amdgcn_wmma_f32_16x16x32_f16(false, as, false, b1,
                                                    (short)0, s1, false, false);
        c1 = __builtin_amdgcn_wmma_f32_16x16x32_f16(false, ac, false, b1,
                                                    (short)0, c1, false, false);
    }

    // C/D layout: lane n = lane&15; VGPR r holds row M = r + 8*hi
    const int ncol = npair * 32 + (lane & 15);
#pragma unroll
    for (int r = 0; r < 8; ++r) {
        int m = mtile * 16 + r + 8 * hi;
        if (m < MBINS) {
            float p0 = s0[r] * s0[r] + c0[r] * c0[r];
            float p1 = s1[r] * s1[r] + c1[r] * c1[r];
            out[(size_t)m * NFRAMES + ncol]      = p0;
            out[(size_t)m * NFRAMES + ncol + 16] = p1;
        }
    }
}

// ---------------- launcher ----------------
extern "C" void kernel_launch(void* const* d_in, const int* in_sizes, int n_in,
                              void* d_out, int out_size, void* d_ws, size_t ws_size,
                              hipStream_t stream)
{
    const float* x    = (const float*)d_in[0];   // [T]
    const float* wsin = (const float*)d_in[1];   // [1025,1,2048]
    const float* wcos = (const float*)d_in[2];   // [1025,1,2048]
    float* out = (float*)d_out;                  // [1,1025,20672]

    // workspace layout (256B-aligned sections), total ~29.7 MB
    size_t off = 0;
    _Float16* xh = (_Float16*)((char*)d_ws + off);
    off += (size_t)T_PAD * sizeof(_Float16);
    off = (off + 255) & ~(size_t)255;
    _Float16* whs = (_Float16*)((char*)d_ws + off);
    off += (size_t)MPAD * KDIM * sizeof(_Float16);
    off = (off + 255) & ~(size_t)255;
    _Float16* whc = (_Float16*)((char*)d_ws + off);

    // 1) reflect pad + f16 convert of x
    {
        int nthreads = T_PAD;
        int blocks = (nthreads + 255) / 256;
        pad_convert_kernel<<<blocks, 256, 0, stream>>>(x, xh);
    }
    // 2) weight conversion (M padded to 1040 with zeros)
    {
        int nthreads = MPAD * KDIM;
        int blocks = (nthreads + 255) / 256;
        wconv_kernel<<<blocks, 256, 0, stream>>>(wsin, wcos, whs, whc);
    }
    // 3) WMMA GEMM + power epilogue: 65*646 = 41,990 waves, 8 waves/block
    {
        int total_waves = MTILES * NPAIRS;
        int blocks = (total_waves + 7) / 8;
        spect_wmma_kernel<<<blocks, 256, 0, stream>>>(whs, whc, xh, out);
    }
}
